// BASICVRNN_79594333929660
// MI455X (gfx1250) — matrix operand order repaired and checked by reading, not compiled, not run on hardware
//
#include <hip/hip_runtime.h>
#include <hip/hip_bf16.h>

typedef _Float16 h16_t;
typedef __attribute__((ext_vector_type(16))) _Float16 v16h;
typedef __attribute__((ext_vector_type(8)))  _Float16 v8h;
typedef __attribute__((ext_vector_type(8)))  float    v8f;

#define HB   64
#define TT   256
#define HH   1024
#define HH2  2048
#define HH3  3072

// ---------------- elementwise / conversion kernels ----------------

__global__ void k_f32_to_f16(const float* __restrict__ s, h16_t* __restrict__ d, long n) {
    long i = (long)blockIdx.x * blockDim.x + threadIdx.x;
    if (i < n) d[i] = (h16_t)s[i];
}

// w is (K,N) row-major f32; wt is (N,K) row-major f16 (transposed). Writes coalesced.
__global__ void k_w_transpose_f16(const float* __restrict__ w, h16_t* __restrict__ wt, int K, int N) {
    long i = (long)blockIdx.x * blockDim.x + threadIdx.x;
    if (i < (long)K * N) {
        int n = (int)(i / K);
        int k = (int)(i % K);
        wt[i] = (h16_t)w[(long)k * N + n];
    }
}

__global__ void k_init(const float* __restrict__ h0, float* __restrict__ h,
                       h16_t* __restrict__ h16, float* __restrict__ kld, int n, int Bn) {
    long i = (long)blockIdx.x * blockDim.x + threadIdx.x;
    if (i < n) { float v = h0[i]; h[i] = v; h16[i] = (h16_t)v; }
    if (i < Bn) kld[i] = 0.f;
}

__device__ __forceinline__ float softplus_f(float x) {
    return x > 20.f ? x : log1pf(expf(x));
}

// per batch row: z = mu + sqrt(softplus(sd))*eps ; KL accumulated deterministically
__global__ void k_z_kl(const float* __restrict__ emu, const float* __restrict__ esdl,
                       const float* __restrict__ pmu, const float* __restrict__ psdl,
                       const float* __restrict__ eps, h16_t* __restrict__ z16,
                       float* __restrict__ kld) {
    __shared__ float red[256];
    int b = blockIdx.x;
    float s = 0.f;
    for (int j = threadIdx.x; j < HH; j += blockDim.x) {
        long i = (long)b * HH + j;
        float mu = emu[i];
        float sd = softplus_f(esdl[i]);
        float pm = pmu[i];
        float ps = softplus_f(psdl[i]);
        float z  = mu + sqrtf(sd) * eps[i];
        z16[i] = (h16_t)z;
        float d = pm - mu;
        s += 1.f + (sd - ps) - d * d * expf(-ps) - expf(sd - ps);
    }
    red[threadIdx.x] = s;
    __syncthreads();
    for (int o = 128; o > 0; o >>= 1) {
        if (threadIdx.x < o) red[threadIdx.x] += red[threadIdx.x + o];
        __syncthreads();
    }
    if (threadIdx.x == 0) kld[b] += -0.5f * red[0];
}

__global__ void k_gru(const float* __restrict__ mx, const float* __restrict__ mh,
                      float* __restrict__ h, h16_t* __restrict__ h16,
                      const unsigned char* __restrict__ mask, int t) {
    int b = blockIdx.x;
    bool mt = mask[(long)b * TT + t] != 0;
    long i3 = (long)b * HH3;
    for (int j = threadIdx.x; j < HH; j += blockDim.x) {
        long i = (long)b * HH + j;
        float xz = mx[i3 + j], xr = mx[i3 + HH + j], xh = mx[i3 + 2 * HH + j];
        float hz = mh[i3 + j], hr = mh[i3 + HH + j], hh = mh[i3 + 2 * HH + j];
        float zg = 1.f / (1.f + expf(-(xz + hz)));
        float rg = 1.f / (1.f + expf(-(xr + hr)));
        float cg = tanhf(xh + rg * hh);
        float hv = h[i];
        float hn = zg * hv + (1.f - zg) * cg;
        if (!mt) hn = hv;
        h[i]   = hn;
        h16[i] = (h16_t)hn;
    }
}

__global__ void k_final(const float* __restrict__ h, const float* __restrict__ kld,
                        float* __restrict__ out, int n, int Bn) {
    long i = (long)blockIdx.x * blockDim.x + threadIdx.x;
    if (i < n) { out[i] = h[i]; out[n + i] = h[i]; }
    if (i < Bn) out[2L * n + i] = kld[i];
}

// ---------------- WMMA GEMM ----------------
// C[M,N] = act(concatK(A0,A1)[M,K] @ W[K,N] + bias)
// A rows are f16 with per-phase row stride; columns kc < Ksplit come from A0,
// kc >= Ksplit from A1 (uniform switch -> no divergence). WT is (N,K) f16.
// One wave (32 threads) per workgroup; the wave owns a 16-wide N strip and all
// four 16-row M tiles of a 64-row M block: the B fragment (weights) is loaded
// exactly ONCE per column strip (weights are the bound resource), A is tiny
// and WGP$-resident. Per K-chunk: 2 b128 (B) + 8 b128 (A) + 4 WMMA.

struct GemmDesc {
    const h16_t* A0; long lda0;
    const h16_t* A1; long lda1;
    int Ksplit;          // K elements taken from A0 (multiple of 32)
    const h16_t* WT;     // (N,K)
    const float* bias;   // len N
    void* C; int ldc;
    int K;
};

__device__ __forceinline__ v16h ld_afrag(const h16_t* p) {
    v16h a;
    ((v8h*)&a)[0] = *(const v8h*)(p);
    ((v8h*)&a)[1] = *(const v8h*)(p + 16);
    return a;
}

__device__ __forceinline__ void kloop4(
    const h16_t* __restrict__ A, long lda, int mrow, int l15, int half,
    const h16_t* __restrict__ &bp, int nchunks,
    v8f& c0, v8f& c1, v8f& c2, v8f& c3)
{
    const h16_t* a0 = A + (long)(mrow + l15) * lda + half * 8;
    const h16_t* a1 = a0 + 16 * lda;
    const h16_t* a2 = a1 + 16 * lda;
    const h16_t* a3 = a2 + 16 * lda;
    for (int i = 0; i < nchunks; ++i) {
        v16h b = ld_afrag(bp);
        v16h a = ld_afrag(a0);
        c0 = __builtin_amdgcn_wmma_f32_16x16x32_f16(false, a, false, b, (short)0, c0, false, false);
        a = ld_afrag(a1);
        c1 = __builtin_amdgcn_wmma_f32_16x16x32_f16(false, a, false, b, (short)0, c1, false, false);
        a = ld_afrag(a2);
        c2 = __builtin_amdgcn_wmma_f32_16x16x32_f16(false, a, false, b, (short)0, c2, false, false);
        a = ld_afrag(a3);
        c3 = __builtin_amdgcn_wmma_f32_16x16x32_f16(false, a, false, b, (short)0, c3, false, false);
        a0 += 32; a1 += 32; a2 += 32; a3 += 32; bp += 32;
    }
}

template <int ACT, int OUT16>
__device__ __forceinline__ void store_tile(const v8f& c, int r0, int col, float bv,
                                           void* C, int ldc) {
#pragma unroll
    for (int i = 0; i < 8; ++i) {
        float v = c[i] + bv;
        if (ACT == 1) v = v > 0.f ? v : 0.f;
        long idx = (long)(r0 + i) * ldc + col;
        if (OUT16) ((h16_t*)C)[idx] = (h16_t)v;
        else       ((float*)C)[idx] = v;
    }
}

template <int ACT, int OUT16>
__device__ __forceinline__ void gemm_body(const GemmDesc d, int nb, int mb) {
    const int lane = threadIdx.x & 31;
    const int half = lane >> 4;
    const int l15  = lane & 15;

    // B frag (32x16 KxN): lane n holds column nb+n; lanes 0-15 K 0..7,16..23, 16-31 K 8..15,24..31
    const h16_t* bp = d.WT + (long)(nb + l15) * d.K + half * 8;

    v8f c0 = {}, c1 = {}, c2 = {}, c3 = {};
    kloop4(d.A0, d.lda0, mb, l15, half, bp, d.Ksplit / 32, c0, c1, c2, c3);
    kloop4(d.A1, d.lda1, mb, l15, half, bp, (d.K - d.Ksplit) / 32, c0, c1, c2, c3);

    const int col = nb + l15;
    const float bv = d.bias[col];
    const int r0 = half * 8;
    store_tile<ACT, OUT16>(c0, mb + r0,      col, bv, d.C, d.ldc);
    store_tile<ACT, OUT16>(c1, mb + 16 + r0, col, bv, d.C, d.ldc);
    store_tile<ACT, OUT16>(c2, mb + 32 + r0, col, bv, d.C, d.ldc);
    store_tile<ACT, OUT16>(c3, mb + 48 + r0, col, bv, d.C, d.ldc);
}

template <int ACT, int OUT16>
__global__ __launch_bounds__(32) void k_gemm(GemmDesc d) {
    gemm_body<ACT, OUT16>(d, blockIdx.x * 16, blockIdx.y * 64);
}

// two independent GEMMs fused in one launch (z selects); both: no act, f32 out
__global__ __launch_bounds__(32) void k_gemm_dual(GemmDesc d0, GemmDesc d1) {
    GemmDesc d = (blockIdx.z == 0) ? d0 : d1;
    gemm_body<0, 0>(d, blockIdx.x * 16, blockIdx.y * 64);
}

// ---------------- host ----------------

extern "C" void kernel_launch(void* const* d_in, const int* in_sizes, int n_in,
                              void* d_out, int out_size, void* d_ws, size_t ws_size,
                              hipStream_t stream) {
    const float* x          = (const float*)d_in[0];
    const unsigned char* mask = (const unsigned char*)d_in[1];
    const float* eps        = (const float*)d_in[2];
    const float* h0         = (const float*)d_in[3];
    const float* phi_x_w1   = (const float*)d_in[4];
    const float* phi_x_b1   = (const float*)d_in[5];
    const float* phi_x_w2   = (const float*)d_in[6];
    const float* phi_x_b2   = (const float*)d_in[7];
    const float* enc_w1     = (const float*)d_in[8];
    const float* enc_b1     = (const float*)d_in[9];
    const float* enc_w2     = (const float*)d_in[10];
    const float* enc_b2     = (const float*)d_in[11];
    const float* enc_mean_w = (const float*)d_in[12];
    const float* enc_mean_b = (const float*)d_in[13];
    const float* enc_std_w  = (const float*)d_in[14];
    const float* enc_std_b  = (const float*)d_in[15];
    const float* prior_w1   = (const float*)d_in[16];
    const float* prior_b1   = (const float*)d_in[17];
    const float* prior_w2   = (const float*)d_in[18];
    const float* prior_b2   = (const float*)d_in[19];
    const float* prior_mean_w = (const float*)d_in[20];
    const float* prior_mean_b = (const float*)d_in[21];
    const float* prior_std_w  = (const float*)d_in[22];
    const float* prior_std_b  = (const float*)d_in[23];
    const float* phi_z_w    = (const float*)d_in[24];
    const float* phi_z_b    = (const float*)d_in[25];
    const float* gru_kernel = (const float*)d_in[26];
    const float* gru_rec    = (const float*)d_in[27];
    const float* gru_bias   = (const float*)d_in[28];
    float* out = (float*)d_out;

    // bump allocator on workspace
    char* wsp = (char*)d_ws;
    auto alloc = [&](size_t bytes) -> void* {
        void* p = (void*)wsp;
        wsp += (bytes + 255) & ~(size_t)255;
        return p;
    };

    // transposed f16 weights (N,K)
    h16_t* wt_phi1 = (h16_t*)alloc((size_t)HH  * HH  * 2);
    h16_t* wt_phi2 = (h16_t*)alloc((size_t)HH  * HH  * 2);
    h16_t* wt_enc1 = (h16_t*)alloc((size_t)HH2 * HH2 * 2);
    h16_t* wt_enc2 = (h16_t*)alloc((size_t)HH2 * HH2 * 2);
    h16_t* wt_encm = (h16_t*)alloc((size_t)HH  * HH2 * 2);
    h16_t* wt_encs = (h16_t*)alloc((size_t)HH  * HH2 * 2);
    h16_t* wt_pr1  = (h16_t*)alloc((size_t)HH  * HH  * 2);
    h16_t* wt_pr2  = (h16_t*)alloc((size_t)HH  * HH  * 2);
    h16_t* wt_prm  = (h16_t*)alloc((size_t)HH  * HH  * 2);
    h16_t* wt_prs  = (h16_t*)alloc((size_t)HH  * HH  * 2);
    h16_t* wt_phiz = (h16_t*)alloc((size_t)HH  * HH  * 2);
    h16_t* wt_gruk = (h16_t*)alloc((size_t)HH3 * HH2 * 2);
    h16_t* wt_grur = (h16_t*)alloc((size_t)HH3 * HH  * 2);

    const long NX = (long)HB * TT * HH;
    h16_t* x16   = (h16_t*)alloc((size_t)NX * 2);  // x f16, later reused as phi_x
    h16_t* tmp16 = (h16_t*)alloc((size_t)NX * 2);  // phi_x hidden

    h16_t* enc_h1  = (h16_t*)alloc((size_t)HB * HH2 * 2);
    h16_t* enc_t   = (h16_t*)alloc((size_t)HB * HH2 * 2);
    h16_t* prior_h = (h16_t*)alloc((size_t)HB * HH * 2);
    h16_t* prior_t = (h16_t*)alloc((size_t)HB * HH * 2);
    h16_t* z16     = (h16_t*)alloc((size_t)HB * HH * 2);
    h16_t* phiz16  = (h16_t*)alloc((size_t)HB * HH * 2);
    float* enc_mu  = (float*)alloc((size_t)HB * HH * 4);
    float* enc_sdl = (float*)alloc((size_t)HB * HH * 4);
    float* pr_mu   = (float*)alloc((size_t)HB * HH * 4);
    float* pr_sdl  = (float*)alloc((size_t)HB * HH * 4);
    float* gmx     = (float*)alloc((size_t)HB * HH3 * 4);
    float* gmh     = (float*)alloc((size_t)HB * HH3 * 4);
    float* hbuf    = (float*)alloc((size_t)HB * HH * 4);
    h16_t* h16buf  = (h16_t*)alloc((size_t)HB * HH * 2);
    float* kld     = (float*)alloc((size_t)HB * 4);
    (void)ws_size; (void)n_in; (void)in_sizes; (void)out_size;

    auto wconv = [&](const float* w, h16_t* wt, int K, int N) {
        long n = (long)K * N;
        k_w_transpose_f16<<<dim3((unsigned)((n + 255) / 256)), 256, 0, stream>>>(w, wt, K, N);
    };
    auto mkdesc = [&](const h16_t* A0, long lda0, const h16_t* A1, long lda1, int Ksplit,
                      const h16_t* WT, const float* bias, void* C, int ldc, int K) {
        GemmDesc d; d.A0 = A0; d.lda0 = lda0; d.A1 = A1; d.lda1 = lda1; d.Ksplit = Ksplit;
        d.WT = WT; d.bias = bias; d.C = C; d.ldc = ldc; d.K = K; return d;
    };
    // single GEMM, relu + f16 out
    auto gemm_rf16 = [&](GemmDesc d, int M, int N) {
        dim3 g((unsigned)(N / 16), (unsigned)(M / 64)), blk(32);
        k_gemm<1, 1><<<g, blk, 0, stream>>>(d);
    };
    // fused pair, no act + f32 out (N must match)
    auto gemm_dual = [&](GemmDesc d0, GemmDesc d1, int M, int N) {
        dim3 g((unsigned)(N / 16), (unsigned)(M / 64), 2), blk(32);
        k_gemm_dual<<<g, blk, 0, stream>>>(d0, d1);
    };

    // --- one-time weight conversion + transpose to (N,K) f16 ---
    wconv(phi_x_w1, wt_phi1, HH,  HH);
    wconv(phi_x_w2, wt_phi2, HH,  HH);
    wconv(enc_w1,   wt_enc1, HH2, HH2);
    wconv(enc_w2,   wt_enc2, HH2, HH2);
    wconv(enc_mean_w, wt_encm, HH2, HH);
    wconv(enc_std_w,  wt_encs, HH2, HH);
    wconv(prior_w1, wt_pr1, HH, HH);
    wconv(prior_w2, wt_pr2, HH, HH);
    wconv(prior_mean_w, wt_prm, HH, HH);
    wconv(prior_std_w,  wt_prs, HH, HH);
    wconv(phi_z_w,  wt_phiz, HH, HH);
    wconv(gru_kernel, wt_gruk, HH2, HH3);
    wconv(gru_rec,    wt_grur, HH,  HH3);

    // --- hoisted phi_x over all (B,T): two big parallel GEMMs, M = B*T = 16384 ---
    k_f32_to_f16<<<dim3((unsigned)((NX + 255) / 256)), 256, 0, stream>>>(x, x16, NX);
    gemm_rf16(mkdesc(x16,   HH, x16,   HH, HH, wt_phi1, phi_x_b1, tmp16, HH, HH), HB * TT, HH);
    gemm_rf16(mkdesc(tmp16, HH, tmp16, HH, HH, wt_phi2, phi_x_b2, x16,   HH, HH), HB * TT, HH);
    // x16 now holds phi_x, laid out (B,T,H)

    // --- init state ---
    k_init<<<dim3((HB * HH + 255) / 256), 256, 0, stream>>>(h0, hbuf, h16buf, kld, HB * HH, HB);

    const long ldphix = (long)TT * HH;  // row stride of phi_x over batch

    // --- sequential scan: M=64 GEMMs against L2-resident f16 weights ---
    for (int t = 0; t < TT; ++t) {
        const h16_t* phix_t = x16 + (long)t * HH;  // rows: b -> phix_t + b*ldphix

        // enc FFN; A = [phi_x_t | h] via split-A (no pack kernel)
        gemm_rf16(mkdesc(phix_t, ldphix, h16buf, HH, HH, wt_enc1, enc_b1, enc_h1, HH2, HH2), HB, HH2);
        gemm_rf16(mkdesc(enc_h1, HH2, enc_h1, HH2, HH2, wt_enc2, enc_b2, enc_t, HH2, HH2), HB, HH2);
        gemm_dual(mkdesc(enc_t, HH2, enc_t, HH2, HH2, wt_encm, enc_mean_b, enc_mu,  HH, HH2),
                  mkdesc(enc_t, HH2, enc_t, HH2, HH2, wt_encs, enc_std_b,  enc_sdl, HH, HH2),
                  HB, HH);

        // prior FFN on h
        gemm_rf16(mkdesc(h16buf,  HH, h16buf,  HH, HH, wt_pr1, prior_b1, prior_h, HH, HH), HB, HH);
        gemm_rf16(mkdesc(prior_h, HH, prior_h, HH, HH, wt_pr2, prior_b2, prior_t, HH, HH), HB, HH);
        gemm_dual(mkdesc(prior_t, HH, prior_t, HH, HH, wt_prm, prior_mean_b, pr_mu,  HH, HH),
                  mkdesc(prior_t, HH, prior_t, HH, HH, wt_prs, prior_std_b,  pr_sdl, HH, HH),
                  HB, HH);

        k_z_kl<<<dim3(HB), 256, 0, stream>>>(enc_mu, enc_sdl, pr_mu, pr_sdl, eps, z16, kld);

        gemm_rf16(mkdesc(z16, HH, z16, HH, HH, wt_phiz, phi_z_b, phiz16, HH, HH), HB, HH);

        // GRU: mx uses A = [phi_x_t | phi_z] via split-A; mh uses h
        gemm_dual(mkdesc(phix_t, ldphix, phiz16, HH, HH, wt_gruk, gru_bias,       gmx, HH3, HH2),
                  mkdesc(h16buf, HH,     h16buf, HH, HH, wt_grur, gru_bias + HH3, gmh, HH3, HH),
                  HB, HH3);

        k_gru<<<dim3(HB), 256, 0, stream>>>(gmx, gmh, hbuf, h16buf, mask, t);
    }

    k_final<<<dim3((HB * HH + 255) / 256), 256, 0, stream>>>(hbuf, kld, out, HB * HH, HB);
}
